// TemporalCommunityGNN_85255100826173
// MI455X (gfx1250) — matrix operand order repaired and checked by reading, not compiled, not run on hardware
//
#include <hip/hip_runtime.h>

// ---------------------------------------------------------------- types
typedef __attribute__((ext_vector_type(16))) _Float16 v16h;
typedef __attribute__((ext_vector_type(8)))  float    v8f;
typedef __attribute__((ext_vector_type(4)))  unsigned int v4u;

union FragA { v16h h; v4u q[2]; };

// ---------------------------------------------------------------- sizes
constexpr int T_   = 12;     // months
constexpr int NU_  = 10000;  // users
constexpr int NT_  = 2500;   // tags
constexpr int FU_  = 64;
constexpr int FT_  = 32;
constexpr int H_   = 128;
constexpr int D_   = 256;
constexpr int FFN_ = 256;
constexpr int S_   = 12;     // sequence length
constexpr int SP_  = 16;     // padded sequence length
constexpr int ETT_ = 50000;
constexpr int EUT_ = 100000;
constexpr int ETU_ = 100000;

enum { GF_ACC = 1, GF_BIAS = 2, GF_RELU = 4 };

// ---------------------------------------------------------------- WMMA GEMM
// C[M,N] = A[M,K](f16,row-major) @ Bp (packed f16 fragments) (+bias)(+C)(relu)
// One wave per 16x64 output strip (4 N-tiles): one A fragment feeds 4 WMMAs.
// N must be a multiple of 64. Batch over gridDim.z.
__global__ __launch_bounds__(32)
void gemm16(const _Float16* __restrict__ A, long long sA,
            const _Float16* __restrict__ Bp, long long sB,
            const float* __restrict__ bias,
            float* __restrict__ C, long long sC,
            int M, int N, int K, int lda, int ldc, int flags)
{
  const int lane = threadIdx.x;
  const int nt0  = blockIdx.x * 4;      // first of 4 N-tiles
  const int m0   = blockIdx.y * 16;
  const int b    = blockIdx.z;
  A  += (long long)b * sA;
  Bp += (long long)b * sB;
  C  += (long long)b * sC;

  const int half = lane >> 4;           // 0: K{0..7,16..23}  1: K{8..15,24..31}
  const int l15  = lane & 15;
  const int arow = m0 + l15;
  const int ntn  = N >> 4;

  const _Float16* Ar = A + (long long)arow * lda + (half << 3);
  const _Float16* Bb = Bp + ((long long)nt0 << 9) + (lane << 4);
  const long long bstep = (long long)ntn << 9;   // halfs per K-tile row of fragments

  v8f acc0 = {}, acc1 = {}, acc2 = {}, acc3 = {};

  if ((m0 + 16) <= M) {
    // ---- full tile: no predication anywhere in the hot loop
    for (int k0 = 0; k0 < K; k0 += 32) {
      const v4u* p = (const v4u*)(Ar + k0);
      FragA a;
      a.q[0] = p[0];   // K offsets +0..7   (or +8..15)
      a.q[1] = p[2];   // K offsets +16..23 (or +24..31)
      __builtin_prefetch(Ar + k0 + 32, 0, 1);
      const _Float16* bb = Bb + (long long)(k0 >> 5) * bstep;
      const v16h b0 = *(const v16h*)(bb);
      const v16h b1 = *(const v16h*)(bb + 512);
      const v16h b2 = *(const v16h*)(bb + 1024);
      const v16h b3 = *(const v16h*)(bb + 1536);
      acc0 = __builtin_amdgcn_wmma_f32_16x16x32_f16(false, a.h, false, b0, (short)0, acc0, false, false);
      acc1 = __builtin_amdgcn_wmma_f32_16x16x32_f16(false, a.h, false, b1, (short)0, acc1, false, false);
      acc2 = __builtin_amdgcn_wmma_f32_16x16x32_f16(false, a.h, false, b2, (short)0, acc2, false, false);
      acc3 = __builtin_amdgcn_wmma_f32_16x16x32_f16(false, a.h, false, b3, (short)0, acc3, false, false);
    }
  } else {
    // ---- partial last M-tile: per-lane row guard, zero-filled A rows
    const bool arow_ok = arow < M;
    for (int k0 = 0; k0 < K; k0 += 32) {
      FragA a;
      v4u z = {}; a.q[0] = z; a.q[1] = z;
      if (arow_ok) {
        const v4u* p = (const v4u*)(Ar + k0);
        a.q[0] = p[0];
        a.q[1] = p[2];
      }
      const _Float16* bb = Bb + (long long)(k0 >> 5) * bstep;
      const v16h b0 = *(const v16h*)(bb);
      const v16h b1 = *(const v16h*)(bb + 512);
      const v16h b2 = *(const v16h*)(bb + 1024);
      const v16h b3 = *(const v16h*)(bb + 1536);
      acc0 = __builtin_amdgcn_wmma_f32_16x16x32_f16(false, a.h, false, b0, (short)0, acc0, false, false);
      acc1 = __builtin_amdgcn_wmma_f32_16x16x32_f16(false, a.h, false, b1, (short)0, acc1, false, false);
      acc2 = __builtin_amdgcn_wmma_f32_16x16x32_f16(false, a.h, false, b2, (short)0, acc2, false, false);
      acc3 = __builtin_amdgcn_wmma_f32_16x16x32_f16(false, a.h, false, b3, (short)0, acc3, false, false);
    }
  }

  v8f accs[4] = { acc0, acc1, acc2, acc3 };
#pragma unroll
  for (int t = 0; t < 4; ++t) {
    const int col = (nt0 + t) * 16 + l15;
    const float bv = (flags & GF_BIAS) ? bias[col] : 0.0f;
#pragma unroll
    for (int r = 0; r < 8; ++r) {
      int row = m0 + r + half * 8;
      if (row < M) {
        float v = accs[t][r] + bv;
        float* cp = C + (long long)row * ldc + col;
        if (flags & GF_ACC)  v += *cp;
        if (flags & GF_RELU) v = v > 0.0f ? v : 0.0f;
        *cp = v;
      }
    }
  }
}

// ---------------------------------------------------------------- weight pack
// B[batch][K][N] f32  ->  Bp[batch][K/32][N/16][lane 0..31][16] f16
__global__ void pack_b(const float* __restrict__ B, _Float16* __restrict__ Bp,
                       int K, int N, long long total)
{
  long long i = (long long)blockIdx.x * blockDim.x + threadIdx.x;
  if (i >= total) return;
  int e    = i & 15;
  int lane = (int)((i >> 4) & 31);
  long long t = i >> 9;
  int ntn = N >> 4;
  int nt  = (int)(t % ntn);  long long t2 = t / ntn;
  int kt  = (int)(t2 % (K >> 5));
  long long bidx = t2 / (K >> 5);
  int col = nt * 16 + (lane & 15);
  int row = kt * 32 + ((lane >> 4) << 3) + (e & 7) + ((e >> 3) << 4);
  Bp[i] = (_Float16)B[bidx * (long long)K * N + (long long)row * N + col];
}

// ---------------------------------------------------------------- BN
__global__ void bn_stats(const float* __restrict__ x, int Nn, int F,
                         float* __restrict__ mean, float* __restrict__ istd)
{
  int f = blockIdx.x, m = blockIdx.y, tid = threadIdx.x;
  x += (long long)m * Nn * F;
  float s = 0.f, s2 = 0.f;
  for (int i = tid; i < Nn; i += blockDim.x) {
    float v = x[(long long)i * F + f];
    s += v; s2 += v * v;
  }
  __shared__ float sh[256], sh2[256];
  sh[tid] = s; sh2[tid] = s2; __syncthreads();
  for (int sft = 128; sft > 0; sft >>= 1) {
    if (tid < sft) { sh[tid] += sh[tid + sft]; sh2[tid] += sh2[tid + sft]; }
    __syncthreads();
  }
  if (tid == 0) {
    float mu  = sh[0] / Nn;
    float var = sh2[0] / Nn - mu * mu;
    mean[m * F + f] = mu;
    istd[m * F + f] = rsqrtf(var + 1e-5f);
  }
}

__global__ void bn_apply(const float* __restrict__ x,
                         const float* __restrict__ mean, const float* __restrict__ istd,
                         const float* __restrict__ g, const float* __restrict__ b,
                         _Float16* __restrict__ out, int Nn, int F)
{
  int m = blockIdx.y;
  long long i = (long long)blockIdx.x * blockDim.x + threadIdx.x;
  if (i >= (long long)Nn * F) return;
  int f = (int)(i % F);
  float v = (x[(long long)m * Nn * F + i] - mean[m * F + f]) * istd[m * F + f]
            * g[f] + b[f];
  out[(long long)m * Nn * F + i] = (_Float16)v;
}

// ---------------------------------------------------------------- utility
__global__ void fillf(float* __restrict__ p, long long n, float v)
{
  long long i = (long long)blockIdx.x * blockDim.x + threadIdx.x;
  if (i < n) p[i] = v;
}

__global__ void cvt_h(const float* __restrict__ s, _Float16* __restrict__ d, long long n)
{
  long long i = (long long)blockIdx.x * blockDim.x + threadIdx.x;
  if (i < n) d[i] = (_Float16)s[i];
}

__global__ void cvt_pad(const float* __restrict__ s, _Float16* __restrict__ d,
                        int rows, int rows_pad, int cols)
{
  int i = blockIdx.x * blockDim.x + threadIdx.x;
  if (i >= rows_pad * cols) return;
  d[i] = (i / cols < rows) ? (_Float16)s[i] : (_Float16)0.0f;
}

// ---------------------------------------------------------------- scatter-mean
__global__ void scatter_add(const _Float16* __restrict__ xsrc, long long sx,
                            const int* __restrict__ src, const int* __restrict__ dst,
                            long long se,
                            float* __restrict__ agg, long long sagg,
                            float* __restrict__ cnt, long long scnt,
                            int E, int Hf)
{
  int m = blockIdx.y;
  xsrc += (long long)m * sx;  src += (long long)m * se;  dst += (long long)m * se;
  agg  += (long long)m * sagg; cnt += (long long)m * scnt;
  int chunks = Hf >> 2;
  long long i = (long long)blockIdx.x * blockDim.x + threadIdx.x;
  if (i >= (long long)E * chunks) return;
  int e = (int)(i / chunks);
  int c = (int)(i % chunks) << 2;
  int s = src[e], d = dst[e];
  const _Float16* xp = xsrc + (long long)s * Hf + c;
  float* ap = agg + (long long)d * Hf + c;
  atomicAdd(ap + 0, (float)xp[0]);
  atomicAdd(ap + 1, (float)xp[1]);
  atomicAdd(ap + 2, (float)xp[2]);
  atomicAdd(ap + 3, (float)xp[3]);
  if (c == 0) atomicAdd(cnt + d, 1.0f);
}

__global__ void agg_fin(const float* __restrict__ agg, const float* __restrict__ cnt,
                        _Float16* __restrict__ out, long long sagg, long long scnt,
                        int Nn, int Hf)
{
  int m = blockIdx.y;
  long long i = (long long)blockIdx.x * blockDim.x + threadIdx.x;
  if (i >= (long long)Nn * Hf) return;
  int n = (int)(i / Hf);
  float cv = cnt[(long long)m * scnt + n];
  cv = cv > 1.0f ? cv : 1.0f;
  out[(long long)m * sagg + i] = (_Float16)(agg[(long long)m * sagg + i] / cv);
}

__global__ void combine2(const float* __restrict__ a, const float* __restrict__ b,
                         float scale, _Float16* __restrict__ xh, long long n)
{
  long long i = (long long)blockIdx.x * blockDim.x + threadIdx.x;
  if (i >= n) return;
  float v = a[i];
  if (b) v += b[i];
  v *= scale;
  v = v > 0.0f ? v : 0.0f;
  xh[i] = (_Float16)v;
}

// ---------------------------------------------------------------- month mean
__global__ void month_mean(const _Float16* __restrict__ xu,
                           const _Float16* __restrict__ xt, float* __restrict__ emb)
{
  int f = blockIdx.x, m = blockIdx.y, tid = threadIdx.x;
  const _Float16* x; int Nn; int col;
  if (f < H_) { x = xu + (long long)m * NU_ * H_; Nn = NU_; col = f; }
  else        { x = xt + (long long)m * NT_ * H_; Nn = NT_; col = f - H_; }
  float s = 0.f;
  for (int i = tid; i < Nn; i += blockDim.x) s += (float)x[(long long)i * H_ + col];
  __shared__ float sh[256];
  sh[tid] = s; __syncthreads();
  for (int sft = 128; sft > 0; sft >>= 1) {
    if (tid < sft) sh[tid] += sh[tid + sft];
    __syncthreads();
  }
  if (tid == 0) emb[m * D_ + f] = sh[0] / Nn;
}

// ---------------------------------------------------------------- attention (tiny)
__global__ void attn_small(const float* __restrict__ q, const float* __restrict__ k,
                           const float* __restrict__ v, float* __restrict__ o)
{
  int t = threadIdx.x;
  if (t >= 4 * S_) return;       // NH*S = 48 work items
  int h = t / S_, i = t % S_;
  const float* qr = q + i * D_ + h * 64;
  float s[S_]; float mx = -1e30f;
  for (int j = 0; j < S_; ++j) {
    const float* kr = k + j * D_ + h * 64;
    float d = 0.f;
    for (int c = 0; c < 64; ++c) d += qr[c] * kr[c];
    d *= 0.125f;                 // 1/sqrt(64)
    s[j] = d; mx = fmaxf(mx, d);
  }
  float sum = 0.f;
  for (int j = 0; j < S_; ++j) { s[j] = __expf(s[j] - mx); sum += s[j]; }
  float inv = 1.0f / sum;
  for (int c = 0; c < 64; ++c) {
    float acc = 0.f;
    for (int j = 0; j < S_; ++j) acc += s[j] * v[j * D_ + h * 64 + c];
    o[i * D_ + h * 64 + c] = acc * inv;
  }
}

// ---------------------------------------------------------------- layernorm (post-add)
__global__ void ln_add(float* __restrict__ x, const float* __restrict__ res,
                       const float* __restrict__ g, const float* __restrict__ b,
                       _Float16* __restrict__ xh, int cols)
{
  int r = blockIdx.x, c = threadIdx.x;
  float v = x[r * cols + c] + res[r * cols + c];
  __shared__ float sh[256];
  sh[c] = v; __syncthreads();
  for (int sft = 128; sft > 0; sft >>= 1) {
    if (c < sft) sh[c] += sh[c + sft];
    __syncthreads();
  }
  float mu = sh[0] / cols; __syncthreads();
  float dv = v - mu;
  sh[c] = dv * dv; __syncthreads();
  for (int sft = 128; sft > 0; sft >>= 1) {
    if (c < sft) sh[c] += sh[c + sft];
    __syncthreads();
  }
  float var = sh[0] / cols;
  float y = dv * rsqrtf(var + 1e-5f) * g[c] + b[c];
  x[r * cols + c]  = y;
  xh[r * cols + c] = (_Float16)y;
}

// ---------------------------------------------------------------- head
__global__ void head_k(const float* __restrict__ x, const float* __restrict__ W,
                       const float* __restrict__ bias, float* __restrict__ out)
{
  int j = threadIdx.x;
  if (j >= 4) return;
  float acc = bias[j];
  for (int kx = 0; kx < D_; ++kx) acc += x[kx] * W[kx * 4 + j];
  out[j] = acc;
}

// ---------------------------------------------------------------- host
extern "C" void kernel_launch(void* const* d_in, const int* in_sizes, int n_in,
                              void* d_out, int out_size, void* d_ws, size_t ws_size,
                              hipStream_t stream)
{
  (void)in_sizes; (void)n_in; (void)out_size; (void)ws_size;

  const float* x_user  = (const float*)d_in[0];
  const float* x_tag   = (const float*)d_in[1];
  const int*   e_tt    = (const int*)d_in[2];
  const int*   e_ut_s  = (const int*)d_in[3];
  const int*   e_ut_d  = (const int*)d_in[4];
  const int*   e_tu_s  = (const int*)d_in[5];
  const int*   e_tu_d  = (const int*)d_in[6];
  const float* bn_ug   = (const float*)d_in[7];
  const float* bn_ub   = (const float*)d_in[8];
  const float* bn_tg   = (const float*)d_in[9];
  const float* bn_tb   = (const float*)d_in[10];
  const float* proj_uW = (const float*)d_in[11];
  const float* proj_ub = (const float*)d_in[12];
  const float* proj_tW = (const float*)d_in[13];
  const float* proj_tb = (const float*)d_in[14];
  const float* conv_Wl = (const float*)d_in[15];
  const float* conv_bl = (const float*)d_in[16];
  const float* conv_Wr = (const float*)d_in[17];
  const float* Wq = (const float*)d_in[18]; const float* bq = (const float*)d_in[19];
  const float* Wk = (const float*)d_in[20]; const float* bk = (const float*)d_in[21];
  const float* Wv = (const float*)d_in[22]; const float* bv = (const float*)d_in[23];
  const float* Wo = (const float*)d_in[24]; const float* bo = (const float*)d_in[25];
  const float* ln1g = (const float*)d_in[26]; const float* ln1b = (const float*)d_in[27];
  const float* W1 = (const float*)d_in[28]; const float* b1 = (const float*)d_in[29];
  const float* W2 = (const float*)d_in[30]; const float* b2 = (const float*)d_in[31];
  const float* ln2g = (const float*)d_in[32]; const float* ln2b = (const float*)d_in[33];
  const float* headW = (const float*)d_in[34]; const float* headb = (const float*)d_in[35];
  float* out = (float*)d_out;

  // ---- workspace bump allocator (256B aligned)
  char* ws = (char*)d_ws;
  size_t off = 0;
  auto alloc = [&](size_t bytes) -> char* {
    off = (off + 255) & ~(size_t)255;
    char* p = ws + off;
    off += bytes;
    return p;
  };

  // f32 buffers
  float* agg_tt = (float*)alloc((size_t)T_ * NT_ * H_ * 4);  // also o_tt / proj-tag scratch
  float* agg_ut = (float*)alloc((size_t)T_ * NT_ * H_ * 4);  // also o_ut
  float* agg_tu = (float*)alloc((size_t)T_ * NU_ * H_ * 4);  // also o_tu / proj-user scratch
  float* cnt_tt = (float*)alloc((size_t)T_ * NT_ * 4);
  float* cnt_ut = (float*)alloc((size_t)T_ * NT_ * 4);
  float* cnt_tu = (float*)alloc((size_t)T_ * NU_ * 4);
  float* mu_u   = (float*)alloc((size_t)T_ * FU_ * 4);
  float* is_u   = (float*)alloc((size_t)T_ * FU_ * 4);
  float* mu_t   = (float*)alloc((size_t)T_ * FT_ * 4);
  float* is_t   = (float*)alloc((size_t)T_ * FT_ * 4);
  float* xbuf   = (float*)alloc((size_t)S_ * D_ * 4);
  float* qbuf   = (float*)alloc((size_t)S_ * D_ * 4);
  float* kbuf   = (float*)alloc((size_t)S_ * D_ * 4);
  float* vbuf   = (float*)alloc((size_t)S_ * D_ * 4);
  float* obuf   = (float*)alloc((size_t)S_ * D_ * 4);
  float* tmpbuf = (float*)alloc((size_t)S_ * D_ * 4);
  float* fbuf   = (float*)alloc((size_t)S_ * FFN_ * 4);

  // f16 buffers
  _Float16* xu_h    = (_Float16*)alloc((size_t)T_ * NU_ * H_ * 2);
  _Float16* xt_h    = (_Float16*)alloc((size_t)T_ * NT_ * H_ * 2);
  _Float16* aggh_tt = (_Float16*)alloc((size_t)T_ * NT_ * H_ * 2);
  _Float16* aggh_ut = (_Float16*)alloc((size_t)T_ * NT_ * H_ * 2);
  _Float16* aggh_tu = (_Float16*)alloc((size_t)T_ * NU_ * H_ * 2);
  _Float16* xu_n = aggh_tu;   // alias: used only before conv layer 0
  _Float16* xt_n = aggh_tt;   // alias: used only before conv layer 0
  _Float16* x_h  = (_Float16*)alloc((size_t)SP_ * D_ * 2);
  _Float16* o_h  = (_Float16*)alloc((size_t)SP_ * D_ * 2);
  _Float16* f_h  = (_Float16*)alloc((size_t)SP_ * FFN_ * 2);

  // packed weights (f16 fragment layout); packed element count == K*N
  _Float16* pw_pu = (_Float16*)alloc((size_t)FU_ * H_ * 2);
  _Float16* pw_pt = (_Float16*)alloc((size_t)FT_ * H_ * 2);
  _Float16* pw_Wl = (_Float16*)alloc((size_t)9 * H_ * H_ * 2);
  _Float16* pw_Wr = (_Float16*)alloc((size_t)9 * H_ * H_ * 2);
  _Float16* pw_q  = (_Float16*)alloc((size_t)3 * D_ * D_ * 2);
  _Float16* pw_k  = (_Float16*)alloc((size_t)3 * D_ * D_ * 2);
  _Float16* pw_v  = (_Float16*)alloc((size_t)3 * D_ * D_ * 2);
  _Float16* pw_o  = (_Float16*)alloc((size_t)3 * D_ * D_ * 2);
  _Float16* pw_f1 = (_Float16*)alloc((size_t)3 * D_ * FFN_ * 2);
  _Float16* pw_f2 = (_Float16*)alloc((size_t)3 * FFN_ * D_ * 2);

  auto cd = [](long long a, long long b) -> unsigned { return (unsigned)((a + b - 1) / b); };

  auto pack = [&](const float* B, _Float16* Bp, int K, int N, int batch) {
    long long total = (long long)batch * K * N;
    pack_b<<<cd(total, 256), 256, 0, stream>>>(B, Bp, K, N, total);
  };

  auto gemm = [&](const _Float16* A, long long sA, const _Float16* Bp, long long sB,
                  const float* bias, float* C, long long sC,
                  int M, int N, int K, int lda, int ldc, int flags, int batch) {
    dim3 grid(N / 64, cd(M, 16), batch);   // 4 N-tiles per wave
    gemm16<<<grid, 32, 0, stream>>>(A, sA, Bp, sB, bias, C, sC, M, N, K, lda, ldc, flags);
  };

  // ---- 1. pack all weights into WMMA-fragment layout
  pack(proj_uW, pw_pu, FU_, H_, 1);
  pack(proj_tW, pw_pt, FT_, H_, 1);
  pack(conv_Wl, pw_Wl, H_, H_, 9);
  pack(conv_Wr, pw_Wr, H_, H_, 9);
  pack(Wq, pw_q, D_, D_, 3);
  pack(Wk, pw_k, D_, D_, 3);
  pack(Wv, pw_v, D_, D_, 3);
  pack(Wo, pw_o, D_, D_, 3);
  pack(W1, pw_f1, D_, FFN_, 3);
  pack(W2, pw_f2, FFN_, D_, 3);

  // ---- 2. batchnorm (batch stats per month) + f16 normalized features
  bn_stats<<<dim3(FU_, T_), 256, 0, stream>>>(x_user, NU_, FU_, mu_u, is_u);
  bn_stats<<<dim3(FT_, T_), 256, 0, stream>>>(x_tag,  NT_, FT_, mu_t, is_t);
  bn_apply<<<dim3(cd((long long)NU_ * FU_, 256), T_), 256, 0, stream>>>(
      x_user, mu_u, is_u, bn_ug, bn_ub, xu_n, NU_, FU_);
  bn_apply<<<dim3(cd((long long)NT_ * FT_, 256), T_), 256, 0, stream>>>(
      x_tag, mu_t, is_t, bn_tg, bn_tb, xt_n, NT_, FT_);

  // ---- 3. projections (WMMA GEMM), then f16 node features
  gemm(xu_n, (long long)NU_ * FU_, pw_pu, 0, proj_ub, agg_tu, (long long)NU_ * H_,
       NU_, H_, FU_, FU_, H_, GF_BIAS, T_);
  gemm(xt_n, (long long)NT_ * FT_, pw_pt, 0, proj_tb, agg_tt, (long long)NT_ * H_,
       NT_, H_, FT_, FT_, H_, GF_BIAS, T_);
  cvt_h<<<cd((long long)T_ * NU_ * H_, 256), 256, 0, stream>>>(agg_tu, xu_h, (long long)T_ * NU_ * H_);
  cvt_h<<<cd((long long)T_ * NT_ * H_, 256), 256, 0, stream>>>(agg_tt, xt_h, (long long)T_ * NT_ * H_);

  // ---- 4. three hetero SAGE layers
  for (int l = 0; l < 3; ++l) {
    // zero aggregation buffers + counts
    fillf<<<cd((long long)T_ * NT_ * H_, 256), 256, 0, stream>>>(agg_tt, (long long)T_ * NT_ * H_, 0.f);
    fillf<<<cd((long long)T_ * NT_ * H_, 256), 256, 0, stream>>>(agg_ut, (long long)T_ * NT_ * H_, 0.f);
    fillf<<<cd((long long)T_ * NU_ * H_, 256), 256, 0, stream>>>(agg_tu, (long long)T_ * NU_ * H_, 0.f);
    fillf<<<cd((long long)T_ * NT_, 256), 256, 0, stream>>>(cnt_tt, (long long)T_ * NT_, 0.f);
    fillf<<<cd((long long)T_ * NT_, 256), 256, 0, stream>>>(cnt_ut, (long long)T_ * NT_, 0.f);
    fillf<<<cd((long long)T_ * NU_, 256), 256, 0, stream>>>(cnt_tu, (long long)T_ * NU_, 0.f);

    // scatter-mean (atomic f32 accumulation from f16 features)
    scatter_add<<<dim3(cd((long long)ETT_ * (H_ / 4), 256), T_), 256, 0, stream>>>(
        xt_h, (long long)NT_ * H_, e_tt, e_tt + ETT_, (long long)2 * ETT_,
        agg_tt, (long long)NT_ * H_, cnt_tt, NT_, ETT_, H_);
    scatter_add<<<dim3(cd((long long)EUT_ * (H_ / 4), 256), T_), 256, 0, stream>>>(
        xu_h, (long long)NU_ * H_, e_ut_s, e_ut_d, (long long)EUT_,
        agg_ut, (long long)NT_ * H_, cnt_ut, NT_, EUT_, H_);
    scatter_add<<<dim3(cd((long long)ETU_ * (H_ / 4), 256), T_), 256, 0, stream>>>(
        xt_h, (long long)NT_ * H_, e_tu_s, e_tu_d, (long long)ETU_,
        agg_tu, (long long)NU_ * H_, cnt_tu, NU_, ETU_, H_);

    // divide by count, convert to f16 GEMM operands
    agg_fin<<<dim3(cd((long long)NT_ * H_, 256), T_), 256, 0, stream>>>(
        agg_tt, cnt_tt, aggh_tt, (long long)NT_ * H_, NT_, NT_, H_);
    agg_fin<<<dim3(cd((long long)NT_ * H_, 256), T_), 256, 0, stream>>>(
        agg_ut, cnt_ut, aggh_ut, (long long)NT_ * H_, NT_, NT_, H_);
    agg_fin<<<dim3(cd((long long)NU_ * H_, 256), T_), 256, 0, stream>>>(
        agg_tu, cnt_tu, aggh_tu, (long long)NU_ * H_, NU_, NU_, H_);

    // o_xx = agg @ Wl + bl + x_dst @ Wr   (reuse agg f32 buffers as outputs)
    const long long sT = (long long)NT_ * H_, sU = (long long)NU_ * H_;
    gemm(aggh_tt, sT, pw_Wl + (size_t)(l * 3 + 0) * H_ * H_, 0,
         conv_bl + (l * 3 + 0) * H_, agg_tt, sT, NT_, H_, H_, H_, H_, GF_BIAS, T_);
    gemm(xt_h,    sT, pw_Wr + (size_t)(l * 3 + 0) * H_ * H_, 0,
         nullptr, agg_tt, sT, NT_, H_, H_, H_, H_, GF_ACC, T_);
    gemm(aggh_ut, sT, pw_Wl + (size_t)(l * 3 + 1) * H_ * H_, 0,
         conv_bl + (l * 3 + 1) * H_, agg_ut, sT, NT_, H_, H_, H_, H_, GF_BIAS, T_);
    gemm(xt_h,    sT, pw_Wr + (size_t)(l * 3 + 1) * H_ * H_, 0,
         nullptr, agg_ut, sT, NT_, H_, H_, H_, H_, GF_ACC, T_);
    gemm(aggh_tu, sU, pw_Wl + (size_t)(l * 3 + 2) * H_ * H_, 0,
         conv_bl + (l * 3 + 2) * H_, agg_tu, sU, NU_, H_, H_, H_, H_, GF_BIAS, T_);
    gemm(xu_h,    sU, pw_Wr + (size_t)(l * 3 + 2) * H_ * H_, 0,
         nullptr, agg_tu, sU, NU_, H_, H_, H_, H_, GF_ACC, T_);

    // hetero-mean + relu -> new node features (f16)
    combine2<<<cd((long long)T_ * NT_ * H_, 256), 256, 0, stream>>>(
        agg_tt, agg_ut, 0.5f, xt_h, (long long)T_ * NT_ * H_);
    combine2<<<cd((long long)T_ * NU_ * H_, 256), 256, 0, stream>>>(
        agg_tu, nullptr, 1.0f, xu_h, (long long)T_ * NU_ * H_);
  }

  // ---- 5. month embeddings -> transformer input [12, 256]
  month_mean<<<dim3(D_, T_), 256, 0, stream>>>(xu_h, xt_h, xbuf);
  cvt_pad<<<cd((long long)SP_ * D_, 256), 256, 0, stream>>>(xbuf, x_h, S_, SP_, D_);

  // ---- 6. transformer encoder, 3 post-norm layers
  for (int l = 0; l < 3; ++l) {
    const size_t wofs = (size_t)l * D_ * D_;
    gemm(x_h, 0, pw_q + wofs, 0, bq + l * D_, qbuf, 0, S_, D_, D_, D_, D_, GF_BIAS, 1);
    gemm(x_h, 0, pw_k + wofs, 0, bk + l * D_, kbuf, 0, S_, D_, D_, D_, D_, GF_BIAS, 1);
    gemm(x_h, 0, pw_v + wofs, 0, bv + l * D_, vbuf, 0, S_, D_, D_, D_, D_, GF_BIAS, 1);
    attn_small<<<1, 64, 0, stream>>>(qbuf, kbuf, vbuf, obuf);
    cvt_pad<<<cd((long long)SP_ * D_, 256), 256, 0, stream>>>(obuf, o_h, S_, SP_, D_);
    gemm(o_h, 0, pw_o + wofs, 0, bo + l * D_, tmpbuf, 0, S_, D_, D_, D_, D_, GF_BIAS, 1);
    ln_add<<<S_, D_, 0, stream>>>(xbuf, tmpbuf, ln1g + l * D_, ln1b + l * D_, x_h, D_);
    gemm(x_h, 0, pw_f1 + (size_t)l * D_ * FFN_, 0, b1 + l * FFN_, fbuf, 0,
         S_, FFN_, D_, D_, FFN_, GF_BIAS | GF_RELU, 1);
    cvt_pad<<<cd((long long)SP_ * FFN_, 256), 256, 0, stream>>>(fbuf, f_h, S_, SP_, FFN_);
    gemm(f_h, 0, pw_f2 + (size_t)l * FFN_ * D_, 0, b2 + l * D_, tmpbuf, 0,
         S_, D_, FFN_, FFN_, D_, GF_BIAS, 1);
    ln_add<<<S_, D_, 0, stream>>>(xbuf, tmpbuf, ln2g + l * D_, ln2b + l * D_, x_h, D_);
  }

  // ---- 7. head on last token
  head_k<<<1, 32, 0, stream>>>(xbuf + (S_ - 1) * D_, headW, headb, out);
}
